// SchedulingGNN_27118423507563
// MI455X (gfx1250) — matrix operand order repaired and checked by reading, not compiled, not run on hardware
//
#include <hip/hip_runtime.h>
#include <hip/hip_bf16.h>
#include <math.h>
#include <cstdint>

#define NN 50000
#define NE 300000
#define NG 64
#define HID 256
#define EHID 64

typedef __attribute__((ext_vector_type(16))) __bf16 v16bf;
typedef __attribute__((ext_vector_type(8)))  float  v8f;

// ---- gfx1250 async load-to-LDS path (guarded; falls back to sync loads if absent)
#if defined(__gfx1250__) && __has_builtin(__builtin_amdgcn_global_load_async_to_lds_b128) && \
    __has_builtin(__builtin_amdgcn_s_wait_asynccnt)
#define HAVE_ASYNC_LDS 1
#else
#define HAVE_ASYNC_LDS 0
#endif

#if HAVE_ASYNC_LDS
typedef int v4i_gnu __attribute__((vector_size(16)));   // matches builtin's V4i
typedef __attribute__((address_space(1))) v4i_gnu* v4i_gptr;
typedef __attribute__((address_space(3))) v4i_gnu* v4i_lptr;
__device__ __forceinline__ void async_ld_b128(const void* g, void* l) {
    __builtin_amdgcn_global_load_async_to_lds_b128(
        (v4i_gptr)(uintptr_t)g, (v4i_lptr)(uintptr_t)l, 0, 0);
}
#endif

__device__ __forceinline__ unsigned short f2bf(float x) {
    unsigned u = __float_as_uint(x);
    u += 0x7fffu + ((u >> 16) & 1u);
    return (unsigned short)(u >> 16);
}

__device__ __forceinline__ void atomicMaxF(float* addr, float val) {
    if (val >= 0.f) atomicMax((int*)addr, __float_as_int(val));
    else            atomicMin((unsigned int*)addr, __float_as_uint(val));
}

// ---------------- weight convert + transpose: 8 mats of 256x256 -> bf16 [N,K]
__global__ void conv_wt(const float* W0, const float* W1, const float* W2, const float* W3,
                        const float* W4, const float* W5, const float* W6, const float* W7,
                        unsigned short* dst) {
    int i = blockIdx.x * 256 + threadIdx.x;        // 8*65536 total
    if (i >= 8 * 65536) return;
    const float* Ws[8] = {W0, W1, W2, W3, W4, W5, W6, W7};
    int mat = i >> 16, r = i & 65535, n = r >> 8, k = r & 255;
    dst[i] = f2bf(Ws[mat][k * 256 + n]);
}

// ---------------- generic fill
__global__ void fill_f32(float* p, float v, size_t n) {
    size_t i = (size_t)blockIdx.x * 256 + threadIdx.x;
    if (i < n) p[i] = v;
}

// ---------------- node encoder: h = relu(LN(x@W+b)) ; also bf16 copy
__global__ __launch_bounds__(256) void node_encoder(const float* __restrict__ x,
        const float* __restrict__ W, const float* __restrict__ b,
        const float* __restrict__ g, const float* __restrict__ bt,
        float* __restrict__ h, unsigned short* __restrict__ hb) {
    __shared__ float rs[HID];
    __shared__ float stat[2];
    int i = blockIdx.x, c = threadIdx.x;
    float v = b[c];
#pragma unroll
    for (int j = 0; j < 6; ++j) v = fmaf(x[i * 6 + j], W[j * HID + c], v);
    rs[c] = v; __syncthreads();
    for (int s = HID / 2; s > 0; s >>= 1) { if (c < s) rs[c] += rs[c + s]; __syncthreads(); }
    if (c == 0) stat[0] = rs[0] * (1.f / HID);
    __syncthreads();
    float m = stat[0], d = v - m;
    rs[c] = d * d; __syncthreads();
    for (int s = HID / 2; s > 0; s >>= 1) { if (c < s) rs[c] += rs[c + s]; __syncthreads(); }
    if (c == 0) stat[1] = rs[0] * (1.f / HID);
    __syncthreads();
    float y = d * rsqrtf(stat[1] + 1e-5f) * g[c] + bt[c];
    y = fmaxf(y, 0.f);
    h[(size_t)i * HID + c] = y;
    hb[(size_t)i * HID + c] = f2bf(y);
}

// ---------------- edge encoder: e = relu(attr*W + b)  [E,64] fp32
__global__ void edge_encoder(const float* __restrict__ attr, const float* __restrict__ W,
                             const float* __restrict__ b, float* __restrict__ e) {
    size_t i = (size_t)blockIdx.x * 256 + threadIdx.x;
    if (i >= (size_t)NE * EHID) return;
    size_t j = i >> 6; int c = (int)(i & 63);
    e[i] = fmaxf(fmaf(attr[j], W[c], b[c]), 0.f);
}

// ---------------- bf16 WMMA GEMM: C[M,N] = A(bf16 [M,K]) * Wt^T + bias ; Wt is bf16 [N,K]
// Block tile 64(M) x 128(N), 8 waves, each wave 32x32 via 2x2 WMMA 16x16x32.
// Async path: double-buffered LDS, tile t+1 prefetched via global_load_async_to_lds_b128
// while tile t runs on the matrix pipe; s_wait_asynccnt 3 keeps prefetch in flight.
__global__ __launch_bounds__(256) void wmma_gemm_bf16(
        const unsigned short* __restrict__ A, const unsigned short* __restrict__ Wt,
        const float* __restrict__ bias, float* __restrict__ C, int M, int K, int N) {
    constexpr int LDA = 40, LDB = 40;               // bf16 elems, +8 pad (row = 80B, 16B-aligned)
    __shared__ unsigned short sA[2 * 64 * LDA];
    __shared__ unsigned short sB[2 * 128 * LDB];
    const int tid = threadIdx.x;
    const int lane = tid & 31, wave = tid >> 5;
    const int wrow = wave >> 2, wcol = wave & 3;    // 2 x 4 wave grid
    const int m0 = blockIdx.x * 64, n0 = blockIdx.y * 128;
    const int alr = tid >> 2, alc = (tid & 3) * 8;  // A: 64 rows x 32
    const int blr = tid >> 1, blc = (tid & 1) * 16; // B: 128 rows x 32
    const int kh = lane >> 4, ll = lane & 15;

    v8f acc[2][2] = {{{}, {}}, {{}, {}}};

    auto compute_tile = [&](int buf) {
        const unsigned short* bA = sA + buf * 64 * LDA;
        const unsigned short* bB = sB + buf * 128 * LDB;
        union { v16bf v; unsigned u[8]; } fa[2], fb[2];
#pragma unroll
        for (int ms = 0; ms < 2; ++ms) {            // A 16x32 layout (ISA 7.12.2)
            const int mloc = wrow * 32 + ms * 16 + ll;
#pragma unroll
            for (int v = 0; v < 8; ++v) {
                int kk = ((v < 4) ? 0 : 16) + kh * 8 + (v & 3) * 2;
                fa[ms].u[v] = *(const unsigned*)&bA[mloc * LDA + kk];
            }
        }
#pragma unroll
        for (int ns = 0; ns < 2; ++ns) {            // B 32x16 layout: K = (v<4?0:16) + kh*8 + 2(v&3) + p
            const int nloc = wcol * 32 + ns * 16 + ll;
#pragma unroll
            for (int v = 0; v < 8; ++v) {
                int kk = ((v < 4) ? 0 : 16) + kh * 8 + (v & 3) * 2;
                fb[ns].u[v] = *(const unsigned*)&bB[nloc * LDB + kk];
            }
        }
#pragma unroll
        for (int ms = 0; ms < 2; ++ms)
#pragma unroll
            for (int ns = 0; ns < 2; ++ns)
                acc[ms][ns] = __builtin_amdgcn_wmma_f32_16x16x32_bf16(
                    false, fa[ms].v, false, fb[ns].v, (short)0, acc[ms][ns], false, false);
    };

#if HAVE_ASYNC_LDS
    auto issue_tile = [&](int kt, int buf) {
        unsigned short* bA = sA + buf * 64 * LDA;
        unsigned short* bB = sB + buf * 128 * LDB;
        int gm = m0 + alr;
        if (gm < M) {
            async_ld_b128(A + (size_t)gm * K + kt + alc, &bA[alr * LDA + alc]);
        } else {
            uint4 z = {0u, 0u, 0u, 0u};
            *(uint4*)&bA[alr * LDA + alc] = z;
        }
        const unsigned short* wp = Wt + (size_t)(n0 + blr) * K + kt + blc;
        async_ld_b128(wp,     &bB[blr * LDB + blc]);
        async_ld_b128(wp + 8, &bB[blr * LDB + blc + 8]);
    };
    const int NT = K / 32;
    issue_tile(0, 0);
    for (int t = 0; t < NT; ++t) {
        if (t + 1 < NT) {
            issue_tile((t + 1) * 32, (t + 1) & 1);
            __builtin_amdgcn_s_wait_asynccnt(3);    // wait for tile t only
        } else {
            __builtin_amdgcn_s_wait_asynccnt(0);
        }
        __syncthreads();
        compute_tile(t & 1);
        __syncthreads();
    }
#else
    for (int kt = 0; kt < K; kt += 32) {
        {   // A tile: branch-free masked b128 load + b128 LDS store
            int gm = m0 + alr;
            int gmc = (gm < M) ? gm : (M - 1);
            unsigned msk = (gm < M) ? 0xFFFFFFFFu : 0u;
            uint4 t = *(const uint4*)(A + (size_t)gmc * K + kt + alc);
            t.x &= msk; t.y &= msk; t.z &= msk; t.w &= msk;
            *(uint4*)&sA[alr * LDA + alc] = t;
        }
        {   // B tile: 2 x b128 per thread
            const uint4* sp = (const uint4*)(Wt + (size_t)(n0 + blr) * K + kt + blc);
            uint4 t0 = sp[0], t1 = sp[1];
            *(uint4*)&sB[blr * LDB + blc]     = t0;
            *(uint4*)&sB[blr * LDB + blc + 8] = t1;
        }
        __syncthreads();
        compute_tile(0);
        __syncthreads();
    }
#endif

    // C/D layout: M = r + 8*kh, N = lane&15
#pragma unroll
    for (int ms = 0; ms < 2; ++ms) {
#pragma unroll
        for (int ns = 0; ns < 2; ++ns) {
            int nc = n0 + wcol * 32 + ns * 16 + ll;
            float bv = bias[nc];
#pragma unroll
            for (int r = 0; r < 8; ++r) {
                int mm = m0 + wrow * 32 + ms * 16 + kh * 8 + r;
                if (mm < M) C[(size_t)mm * N + nc] = acc[ms][ns][r] + bv;
            }
        }
    }
}

// ---------------- fused edge kernel: a[e,h] = att . lrelu(xl[src]+xr[dst]+e@We)
__global__ __launch_bounds__(256) void edge_attn(const int* __restrict__ ei,
        const float* __restrict__ ef, const float* __restrict__ We,
        const float* __restrict__ att, const float* __restrict__ xl,
        const float* __restrict__ xr, float* __restrict__ ascore, int heads) {
    __shared__ float sE[8][EHID];
    int tid = threadIdx.x, lane = tid & 31, wave = tid >> 5;
    int e = blockIdx.x * 8 + wave;
    bool valid = e < NE;
    int eidx = valid ? e : 0;
    sE[wave][2 * lane]     = ef[(size_t)eidx * EHID + 2 * lane];
    sE[wave][2 * lane + 1] = ef[(size_t)eidx * EHID + 2 * lane + 1];
    __syncthreads();
    int src = ei[eidx], dst = ei[NE + eidx];
    int c0 = lane * 8;
    const float* xls = xl + (size_t)src * HID + c0;
    const float* xrd = xr + (size_t)dst * HID + c0;
    __builtin_prefetch(xls, 0, 0);
    __builtin_prefetch(xrd, 0, 0);
    float acc[8];
#pragma unroll
    for (int j = 0; j < 8; ++j) acc[j] = xls[j] + xrd[j];
#pragma unroll 4
    for (int k = 0; k < EHID; ++k) {
        float ek = sE[wave][k];
        const float* wr = We + k * HID + c0;
#pragma unroll
        for (int j = 0; j < 8; ++j) acc[j] = fmaf(ek, wr[j], acc[j]);
    }
    float s = 0.f;
#pragma unroll
    for (int j = 0; j < 8; ++j) {
        float m = acc[j];
        m = (m > 0.f) ? m : 0.2f * m;               // leaky_relu 0.2
        s = fmaf(m, att[c0 + j], s);
    }
    int g = 32 / heads;                              // 4 (8 heads) or 32 (1 head)
    for (int off = g >> 1; off > 0; off >>= 1) s += __shfl_down(s, off, g);
    if (valid && (lane % g) == 0) ascore[(size_t)e * heads + lane / g] = s;
}

// ---------------- segment max over dst
__global__ void seg_max(const float* __restrict__ a, const int* __restrict__ ei,
                        float* __restrict__ amax, int heads) {
    size_t i = (size_t)blockIdx.x * 256 + threadIdx.x;
    if (i >= (size_t)NE * heads) return;
    int e = (int)(i / heads), h = (int)(i % heads);
    int dst = ei[NE + e];
    atomicMaxF(&amax[(size_t)dst * heads + h], a[i]);
}

// ---------------- ex = exp(a - amax'), den += ex
__global__ void seg_expden(const float* __restrict__ a, const int* __restrict__ ei,
                           const float* __restrict__ amax, float* __restrict__ ex,
                           float* __restrict__ den, int heads) {
    size_t i = (size_t)blockIdx.x * 256 + threadIdx.x;
    if (i >= (size_t)NE * heads) return;
    int e = (int)(i / heads), h = (int)(i % heads);
    int dst = ei[NE + e];
    float mv = amax[(size_t)dst * heads + h];
    if (!(mv > -1e37f)) mv = 0.f;                   // isfinite -> 0 semantics
    float v = __expf(a[i] - mv);
    ex[i] = v;
    atomicAdd(&den[(size_t)dst * heads + h], v);
}

// ---------------- out[dst] += xl[src] * alpha
__global__ void seg_wsum(const float* __restrict__ ex, const float* __restrict__ den,
                         const int* __restrict__ ei, const float* __restrict__ xl,
                         float* __restrict__ out, int heads) {
    size_t i = (size_t)blockIdx.x * 256 + threadIdx.x;
    if (i >= (size_t)NE * HID) return;
    int e = (int)(i >> 8), c = (int)(i & 255);
    int h = c / (HID / heads);
    int src = ei[e], dst = ei[NE + e];
    float alpha = ex[(size_t)e * heads + h] / (den[(size_t)dst * heads + h] + 1e-16f);
    atomicAdd(&out[(size_t)dst * HID + c], xl[(size_t)src * HID + c] * alpha);
}

// ---------------- post layer: h' = [h +] relu(LN(hn + bias)) ; write fp32 + bf16
__global__ __launch_bounds__(256) void post_layer(const float* __restrict__ hn,
        const float* __restrict__ bias, const float* __restrict__ g, const float* __restrict__ bt,
        float* __restrict__ h, unsigned short* __restrict__ hb, int use_res) {
    __shared__ float rs[HID];
    __shared__ float stat[2];
    int i = blockIdx.x, c = threadIdx.x;
    float v = hn[(size_t)i * HID + c] + bias[c];
    rs[c] = v; __syncthreads();
    for (int s = HID / 2; s > 0; s >>= 1) { if (c < s) rs[c] += rs[c + s]; __syncthreads(); }
    if (c == 0) stat[0] = rs[0] * (1.f / HID);
    __syncthreads();
    float m = stat[0], d = v - m;
    rs[c] = d * d; __syncthreads();
    for (int s = HID / 2; s > 0; s >>= 1) { if (c < s) rs[c] += rs[c + s]; __syncthreads(); }
    if (c == 0) stat[1] = rs[0] * (1.f / HID);
    __syncthreads();
    float y = fmaxf(d * rsqrtf(stat[1] + 1e-5f) * g[c] + bt[c], 0.f);
    float hv = use_res ? (h[(size_t)i * HID + c] + y) : y;
    h[(size_t)i * HID + c] = hv;
    hb[(size_t)i * HID + c] = f2bf(hv);
}

// ---------------- pooling scatter
__global__ void pool_scatter(const float* __restrict__ h, const int* __restrict__ batch,
                             float* __restrict__ sums, float* __restrict__ maxs,
                             float* __restrict__ cnt) {
    size_t i = (size_t)blockIdx.x * 256 + threadIdx.x;
    if (i >= (size_t)NN * HID) return;
    int n = (int)(i >> 8), c = (int)(i & 255);
    int gr = batch[n];
    float v = h[i];
    atomicAdd(&sums[(size_t)gr * HID + c], v);
    atomicMaxF(&maxs[(size_t)gr * HID + c], v);
    if (c == 0) atomicAdd(&cnt[gr], 1.f);
}

__global__ void pool_combine(const float* __restrict__ sums, const float* __restrict__ maxs,
                             const float* __restrict__ cnt, const float* __restrict__ pw,
                             float* __restrict__ xp) {
    int i = blockIdx.x * 256 + threadIdx.x;          // NG*HID
    if (i >= NG * HID) return;
    int gr = i >> 8, c = i & 255;
    float e0 = __expf(pw[0]), e1 = __expf(pw[1]), e2 = __expf(pw[2]);
    float inv = 1.f / (e0 + e1 + e2);
    float s = sums[i];
    float mean = s / fmaxf(cnt[gr], 1.f);
    xp[gr * 768 + c]       = mean * (e0 * inv);
    xp[gr * 768 + 256 + c] = maxs[i] * (e1 * inv);
    xp[gr * 768 + 512 + c] = s * (e2 * inv);
}

// ---------------- regressor MLP
__global__ void mlp_ln(const float* __restrict__ in, const float* __restrict__ W,
                       const float* __restrict__ b, const float* __restrict__ g,
                       const float* __restrict__ bt, float* __restrict__ out, int IN, int OUT) {
    extern __shared__ float rs[];                    // OUT + 2
    int gr = blockIdx.x, c = threadIdx.x;
    float v = b[c];
    for (int k = 0; k < IN; ++k) v = fmaf(in[gr * IN + k], W[k * OUT + c], v);
    float* stat = rs + OUT;
    rs[c] = v; __syncthreads();
    for (int s = OUT >> 1; s > 0; s >>= 1) { if (c < s) rs[c] += rs[c + s]; __syncthreads(); }
    if (c == 0) stat[0] = rs[0] / OUT;
    __syncthreads();
    float m = stat[0], d = v - m;
    rs[c] = d * d; __syncthreads();
    for (int s = OUT >> 1; s > 0; s >>= 1) { if (c < s) rs[c] += rs[c + s]; __syncthreads(); }
    if (c == 0) stat[1] = rs[0] / OUT;
    __syncthreads();
    out[gr * OUT + c] = fmaxf(d * rsqrtf(stat[1] + 1e-5f) * g[c] + bt[c], 0.f);
}

__global__ void mlp_relu(const float* __restrict__ in, const float* __restrict__ W,
                         const float* __restrict__ b, float* __restrict__ out, int IN, int OUT) {
    int gr = blockIdx.x, c = threadIdx.x;
    float v = b[c];
    for (int k = 0; k < IN; ++k) v = fmaf(in[gr * IN + k], W[k * OUT + c], v);
    out[gr * OUT + c] = fmaxf(v, 0.f);
}

__global__ void mlp_final(const float* __restrict__ in, const float* __restrict__ W,
                          const float* __restrict__ b, float* __restrict__ out, int IN) {
    int gr = threadIdx.x;                            // 64 graphs
    if (gr >= NG) return;
    float v = b[0];
    for (int k = 0; k < IN; ++k) v = fmaf(in[gr * IN + k], W[k], v);
    out[gr] = v;
}

// =================================================================
extern "C" void kernel_launch(void* const* d_in, const int* in_sizes, int n_in,
                              void* d_out, int out_size, void* d_ws, size_t ws_size,
                              hipStream_t stream) {
    (void)in_sizes; (void)n_in; (void)out_size; (void)ws_size;
    // ---- input mapping (setup_inputs dict insertion order, depth-first)
    const float* x     = (const float*)d_in[0];
    const int*   ei    = (const int*)d_in[1];
    const float* eattr = (const float*)d_in[2];
    const int*   batch = (const int*)d_in[3];
    const float* enc_W = (const float*)d_in[4];
    const float* enc_b = (const float*)d_in[5];
    const float* enc_g = (const float*)d_in[6];
    const float* enc_bt= (const float*)d_in[7];
    const float* edge_W= (const float*)d_in[8];
    const float* edge_b= (const float*)d_in[9];
    const float *gWl[4], *gbl[4], *gWr[4], *gbr[4], *gWe[4], *gatt[4], *gbias[4];
    for (int l = 0; l < 4; ++l) {
        int b0 = 10 + 7 * l;
        gWl[l]   = (const float*)d_in[b0 + 0];
        gbl[l]   = (const float*)d_in[b0 + 1];
        gWr[l]   = (const float*)d_in[b0 + 2];
        gbr[l]   = (const float*)d_in[b0 + 3];
        gWe[l]   = (const float*)d_in[b0 + 4];
        gatt[l]  = (const float*)d_in[b0 + 5];
        gbias[l] = (const float*)d_in[b0 + 6];
    }
    const float *ln_g[4], *ln_b[4];
    for (int l = 0; l < 4; ++l) { ln_g[l] = (const float*)d_in[38 + l]; ln_b[l] = (const float*)d_in[42 + l]; }
    const float* pool_w = (const float*)d_in[46];
    const float* r1_W = (const float*)d_in[47]; const float* r1_b = (const float*)d_in[48];
    const float* r1_g = (const float*)d_in[49]; const float* r1_bt= (const float*)d_in[50];
    const float* r2_W = (const float*)d_in[51]; const float* r2_b = (const float*)d_in[52];
    const float* r2_g = (const float*)d_in[53]; const float* r2_bt= (const float*)d_in[54];
    const float* r3_W = (const float*)d_in[55]; const float* r3_b = (const float*)d_in[56];
    const float* r4_W = (const float*)d_in[57]; const float* r4_b = (const float*)d_in[58];

    // ---- workspace carve
    char* p = (char*)d_ws;
    auto alloc = [&](size_t bytes) -> void* {
        void* r = (void*)p; p += (bytes + 255) & ~(size_t)255; return r;
    };
    unsigned short* wbuf = (unsigned short*)alloc(8 * 65536 * sizeof(unsigned short));
    float* h   = (float*)alloc((size_t)NN * HID * 4);
    unsigned short* hb = (unsigned short*)alloc((size_t)NN * HID * 2);
    float* xl  = (float*)alloc((size_t)NN * HID * 4);
    float* xr  = (float*)alloc((size_t)NN * HID * 4);
    float* hn  = (float*)alloc((size_t)NN * HID * 4);
    float* ef  = (float*)alloc((size_t)NE * EHID * 4);
    float* asc = (float*)alloc((size_t)NE * 8 * 4);
    float* ex  = (float*)alloc((size_t)NE * 8 * 4);
    float* amx = (float*)alloc((size_t)NN * 8 * 4);
    float* den = (float*)alloc((size_t)NN * 8 * 4);
    float* sums= (float*)alloc((size_t)NG * HID * 4);
    float* maxs= (float*)alloc((size_t)NG * HID * 4);
    float* cnt = (float*)alloc((size_t)NG * 4);
    float* xp  = (float*)alloc((size_t)NG * 768 * 4);
    float* z1  = (float*)alloc((size_t)NG * 512 * 4);
    float* z2  = (float*)alloc((size_t)NG * 256 * 4);
    float* z3  = (float*)alloc((size_t)NG * 128 * 4);

    const int heads_l[4] = {8, 8, 8, 1};
    const float NEGINF = -INFINITY;

    // weights -> bf16 transposed [N,K]: mats {Wl0,Wr0,Wl1,Wr1,Wl2,Wr2,Wl3,Wr3}
    conv_wt<<<(8 * 65536) / 256, 256, 0, stream>>>(gWl[0], gWr[0], gWl[1], gWr[1],
                                                   gWl[2], gWr[2], gWl[3], gWr[3], wbuf);
    node_encoder<<<NN, HID, 0, stream>>>(x, enc_W, enc_b, enc_g, enc_bt, h, hb);
    edge_encoder<<<(int)(((size_t)NE * EHID + 255) / 256), 256, 0, stream>>>(eattr, edge_W, edge_b, ef);

    dim3 ggrid((NN + 63) / 64, HID / 128);
    for (int l = 0; l < 4; ++l) {
        int heads = heads_l[l];
        wmma_gemm_bf16<<<ggrid, 256, 0, stream>>>(hb, wbuf + (size_t)(2 * l) * 65536,     gbl[l], xl, NN, HID, HID);
        wmma_gemm_bf16<<<ggrid, 256, 0, stream>>>(hb, wbuf + (size_t)(2 * l + 1) * 65536, gbr[l], xr, NN, HID, HID);
        edge_attn<<<(NE + 7) / 8, 256, 0, stream>>>(ei, ef, gWe[l], gatt[l], xl, xr, asc, heads);

        size_t nh = (size_t)NN * heads;
        size_t eh = (size_t)NE * heads;
        fill_f32<<<(int)((nh + 255) / 256), 256, 0, stream>>>(amx, NEGINF, nh);
        fill_f32<<<(int)((nh + 255) / 256), 256, 0, stream>>>(den, 0.f, nh);
        fill_f32<<<(int)(((size_t)NN * HID + 255) / 256), 256, 0, stream>>>(hn, 0.f, (size_t)NN * HID);
        seg_max<<<(int)((eh + 255) / 256), 256, 0, stream>>>(asc, ei, amx, heads);
        seg_expden<<<(int)((eh + 255) / 256), 256, 0, stream>>>(asc, ei, amx, ex, den, heads);
        seg_wsum<<<(int)(((size_t)NE * HID + 255) / 256), 256, 0, stream>>>(ex, den, ei, xl, hn, heads);
        post_layer<<<NN, HID, 0, stream>>>(hn, gbias[l], ln_g[l], ln_b[l], h, hb, l > 0 ? 1 : 0);
    }

    // pooling
    fill_f32<<<(NG * HID + 255) / 256, 256, 0, stream>>>(sums, 0.f, (size_t)NG * HID);
    fill_f32<<<(NG * HID + 255) / 256, 256, 0, stream>>>(maxs, NEGINF, (size_t)NG * HID);
    fill_f32<<<1, 256, 0, stream>>>(cnt, 0.f, (size_t)NG);
    pool_scatter<<<(int)(((size_t)NN * HID + 255) / 256), 256, 0, stream>>>(h, batch, sums, maxs, cnt);
    pool_combine<<<(NG * HID + 255) / 256, 256, 0, stream>>>(sums, maxs, cnt, pool_w, xp);

    // regressor
    mlp_ln<<<NG, 512, (512 + 2) * sizeof(float), stream>>>(xp, r1_W, r1_b, r1_g, r1_bt, z1, 768, 512);
    mlp_ln<<<NG, 256, (256 + 2) * sizeof(float), stream>>>(z1, r2_W, r2_b, r2_g, r2_bt, z2, 512, 256);
    mlp_relu<<<NG, 128, 0, stream>>>(z2, r3_W, r3_b, z3, 256, 128);
    mlp_final<<<1, 64, 0, stream>>>(z3, r4_W, r4_b, (float*)d_out, 128);
}
